// PainnMessage_21320217657822
// MI455X (gfx1250) — matrix operand order repaired and checked by reading, not compiled, not run on hardware
//
#include <hip/hip_runtime.h>
#include <hip/hip_bf16.h>
#include <math.h>

#define N_NODES 10000
#define N_EDGES 160000
#define FDIM    256
#define RDIM    20
#define EPB     32      // edges per block (two 16-row M-tiles)

typedef __attribute__((ext_vector_type(16))) _Float16 v16h;
typedef __attribute__((ext_vector_type(8)))  float    v8f;

union FragU { v16h v; float4 q[2]; };

// A-fragment (16x32 f16, M=lane&15) from an LDS row pointer (points at A[row]+kt*32).
// Layout per ISA 7.12.2: lanes 0-15 hold K=0..7,16..23; lanes 16-31 hold K=8..15,24..31.
__device__ inline v16h load_a_frag(const _Float16* Arow, int hi) {
  FragU f;
  f.q[0] = *reinterpret_cast<const float4*>(Arow + hi * 8);
  f.q[1] = *reinterpret_cast<const float4*>(Arow + 16 + hi * 8);
  return f.v;
}

// B-fragment from pre-packed global weights: 512 halves per fragment, lane-major.
__device__ inline v16h load_b_frag(const _Float16* Wp, int frag, int lane) {
  FragU f;
  const float4* p = reinterpret_cast<const float4*>(Wp + (size_t)frag * 512 + lane * 16);
  f.q[0] = p[0];
  f.q[1] = p[1];
  return f.v;
}

__device__ inline v8f wmma_f16(v16h a, v16h b, v8f c) {
  return __builtin_amdgcn_wmma_f32_16x16x32_f16(false, a, false, b, (short)0, c, false, false);
}

// Order-preserving float<->uint transform for atomicMin/Max on floats.
__device__ inline unsigned enc_f(float f) {
  unsigned u = __float_as_uint(f);
  return (u >> 31) ? ~u : (u | 0x80000000u);
}
__device__ inline float dec_f(unsigned u) {
  unsigned b = (u >> 31) ? (u & 0x7FFFFFFFu) : ~u;
  return __uint_as_float(b);
}

// ---------------------------------------------------------------------------
// Pack f32 row-major W[K][N] into f16 WMMA B-fragment order.
// Fragment id = kt*numNt + nt. Within fragment: slot = lane*16 + j,
// k_local = 8*(lane>>4) + j (j<8), 16 + 8*(lane>>4) + j-8 (j>=8), n = lane&15  (matches 16-bit B 32x16 layout).
// ---------------------------------------------------------------------------
__global__ __launch_bounds__(256) void pack_weight_f16(const float* __restrict__ W,
                                                       _Float16* __restrict__ out,
                                                       int K, int N) {
  int s = blockIdx.x * 256 + threadIdx.x;
  int total = K * N;
  if (s >= total) return;
  int numNt = N >> 4;
  int frag  = s >> 9;
  int rem   = s & 511;
  int l     = rem >> 4;
  int j     = rem & 15;
  int kt    = frag / numNt;
  int nt    = frag - kt * numNt;
  int n     = l & 15;
  int k     = kt * 32 + ((l >> 4) << 3) + (j & 7) + ((j >> 3) << 4);
  out[s] = (_Float16)W[(size_t)k * N + nt * 16 + n];
}

// ---------------------------------------------------------------------------
// Degree histogram.
// ---------------------------------------------------------------------------
__global__ __launch_bounds__(256) void deg_kernel(const int* __restrict__ dst,
                                                  float* __restrict__ deg) {
  int e = blockIdx.x * 256 + threadIdx.x;
  if (e < N_EDGES) atomicAdd(&deg[dst[e]], 1.0f);
}

// ---------------------------------------------------------------------------
// sum_i log1p(deg[i]) -> single scalar (divide by N later).
// ---------------------------------------------------------------------------
__global__ __launch_bounds__(256) void avg_log_kernel(const float* __restrict__ deg,
                                                      float* __restrict__ out) {
  __shared__ float red[256];
  int tid = threadIdx.x;
  int i = blockIdx.x * 256 + tid;
  red[tid] = (i < N_NODES) ? log1pf(deg[i]) : 0.0f;
  __syncthreads();
  for (int s = 128; s > 0; s >>= 1) {
    if (tid < s) red[tid] += red[tid + s];
    __syncthreads();
  }
  if (tid == 0) atomicAdd(out, red[0]);
}

// ---------------------------------------------------------------------------
// Fused edge kernel: 32 edges per block (two M-tiles), 8 wave32.
// Each B fragment is reused across both M-tiles -> half the L2 weight traffic
// of a 16-edge tile. Dynamic LDS (64 KB): A(32x768) | m(32x256); t aliases A.
// ---------------------------------------------------------------------------
__global__ __launch_bounds__(256) void edge_kernel(
    const float* __restrict__ x,
    const float* __restrict__ edge_rbf,
    const float* __restrict__ edge_vec,
    const float* __restrict__ W_rbf_emb,
    const float* __restrict__ b_rbf_emb,
    const _Float16* __restrict__ Wp_pre,  const float* __restrict__ b_pre,
    const _Float16* __restrict__ Wp_s1,   const float* __restrict__ b_s1,
    const _Float16* __restrict__ Wp_s2,   const float* __restrict__ b_s2,
    const float* __restrict__ W_rbf_lin,
    const int* __restrict__ src, const int* __restrict__ dst,
    float* __restrict__ sum_ms, float* __restrict__ sumsq_ms,
    unsigned* __restrict__ minU, unsigned* __restrict__ maxU,
    float* __restrict__ out_v) {
  extern __shared__ __align__(16) _Float16 shDyn[];
  _Float16* Abuf = shDyn;                    // 32 x 768 (GEMM1 input)
  _Float16* Mbuf = shDyn + EPB * 768;        // 32 x 256 (GEMM1 out / GEMM2 in)
  _Float16* Tbuf = shDyn;                    // 32 x 256, aliases dead Abuf

  __shared__ float shRbf[EPB][RDIM];
  __shared__ float shEv[EPB][3];
  __shared__ int shSrc[EPB];
  __shared__ int shDst[EPB];

  const int tid  = threadIdx.x;
  const int lane = tid & 31;
  const int wave = tid >> 5;     // 0..7
  const int hi   = lane >> 4;
  const int n    = lane & 15;
  const int e0   = blockIdx.x * EPB;

  // -------- phase 0: indices, rbf, edge_vec --------
  if (tid < EPB) {
    shSrc[tid] = src[e0 + tid];
    shDst[tid] = dst[e0 + tid];
  }
  for (int t = tid; t < EPB * RDIM; t += 256) {
    int row = t / RDIM, r = t - row * RDIM;
    shRbf[row][r] = edge_rbf[(size_t)(e0 + row) * RDIM + r];
  }
  for (int t = tid; t < EPB * 3; t += 256) {
    int row = t / 3, k = t - row * 3;
    shEv[row][k] = edge_vec[(size_t)(e0 + row) * 3 + k];
  }
  __syncthreads();

  // -------- phase 1: build A (gather x + rbf embedding on VALU, K=20) --------
  {
    int row = tid & 31;            // 32 rows, 8 threads per row
    int c0  = (tid >> 5) * 32;     // 32 cols each
    const float* xs = x + (size_t)shSrc[row] * FDIM + c0;
    const float* xd = x + (size_t)shDst[row] * FDIM + c0;
    _Float16* Ar = Abuf + row * 768;
#pragma unroll 4
    for (int c = 0; c < 32; ++c) {
      Ar[c0 + c]       = (_Float16)xs[c];
      Ar[256 + c0 + c] = (_Float16)xd[c];
      float acc = b_rbf_emb[c0 + c];
#pragma unroll
      for (int r = 0; r < RDIM; ++r)
        acc += shRbf[row][r] * W_rbf_emb[r * FDIM + c0 + c];
      Ar[512 + c0 + c] = (_Float16)fmaxf(acc, 0.0f);
    }
  }
  __syncthreads();

  // -------- GEMM1: m = A @ W_pre + b_pre (wave cols 32w..32w+31, 2 M-tiles) ----
  {
    v8f acc00 = {}, acc01 = {}, acc10 = {}, acc11 = {};
    const _Float16* A0 = Abuf + n * 768;          // M-tile 0: rows 0..15
    const _Float16* A1 = Abuf + (16 + n) * 768;   // M-tile 1: rows 16..31
    for (int kt = 0; kt < 24; ++kt) {
      v16h a0 = load_a_frag(A0 + kt * 32, hi);
      v16h a1 = load_a_frag(A1 + kt * 32, hi);
      v16h b0 = load_b_frag(Wp_pre, kt * 16 + 2 * wave,     lane);
      v16h b1 = load_b_frag(Wp_pre, kt * 16 + 2 * wave + 1, lane);
      acc00 = wmma_f16(a0, b0, acc00);
      acc10 = wmma_f16(a1, b0, acc10);
      acc01 = wmma_f16(a0, b1, acc01);
      acc11 = wmma_f16(a1, b1, acc11);
    }
#pragma unroll
    for (int t = 0; t < 2; ++t) {
      int col = wave * 32 + t * 16 + n;
      float bp = b_pre[col];
      v8f aA = t ? acc01 : acc00;
      v8f aB = t ? acc11 : acc10;
#pragma unroll
      for (int r = 0; r < 8; ++r) {
        Mbuf[(r + 8 * hi) * FDIM + col]      = (_Float16)(aA[r] + bp);
        Mbuf[(16 + r + 8 * hi) * FDIM + col] = (_Float16)(aB[r] + bp);
      }
    }
  }
  __syncthreads();   // all A reads done -> Tbuf alias of Abuf now safe

  // -------- GEMM2: t = silu(m @ W_s1 + b_s1) --------
  {
    v8f acc00 = {}, acc01 = {}, acc10 = {}, acc11 = {};
    const _Float16* A0 = Mbuf + n * FDIM;
    const _Float16* A1 = Mbuf + (16 + n) * FDIM;
    for (int kt = 0; kt < 8; ++kt) {
      v16h a0 = load_a_frag(A0 + kt * 32, hi);
      v16h a1 = load_a_frag(A1 + kt * 32, hi);
      v16h b0 = load_b_frag(Wp_s1, kt * 16 + 2 * wave,     lane);
      v16h b1 = load_b_frag(Wp_s1, kt * 16 + 2 * wave + 1, lane);
      acc00 = wmma_f16(a0, b0, acc00);
      acc10 = wmma_f16(a1, b0, acc10);
      acc01 = wmma_f16(a0, b1, acc01);
      acc11 = wmma_f16(a1, b1, acc11);
    }
#pragma unroll
    for (int t = 0; t < 2; ++t) {
      int col = wave * 32 + t * 16 + n;
      float bs = b_s1[col];
      v8f aA = t ? acc01 : acc00;
      v8f aB = t ? acc11 : acc10;
#pragma unroll
      for (int r = 0; r < 8; ++r) {
        float m0 = aA[r] + bs;
        float m1 = aB[r] + bs;
        Tbuf[(r + 8 * hi) * FDIM + col]      = (_Float16)(m0 / (1.0f + __expf(-m0)));
        Tbuf[(16 + r + 8 * hi) * FDIM + col] = (_Float16)(m1 / (1.0f + __expf(-m1)));
      }
    }
  }
  __syncthreads();

  // -------- GEMM3 + epilogue: matched column triples (g_state/g_edge/ms) ------
  for (int g = 0; g < 2; ++g) {
    int cb  = 16 * (wave + 8 * g);
    int ntS = cb >> 4;
    v8f aS0 = {}, aS1 = {}, aE0 = {}, aE1 = {}, aM0 = {}, aM1 = {};
    const _Float16* A0 = Tbuf + n * FDIM;
    const _Float16* A1 = Tbuf + (16 + n) * FDIM;
    for (int kt = 0; kt < 8; ++kt) {
      v16h a0 = load_a_frag(A0 + kt * 32, hi);
      v16h a1 = load_a_frag(A1 + kt * 32, hi);
      v16h bS = load_b_frag(Wp_s2, kt * 48 + ntS,      lane);
      v16h bE = load_b_frag(Wp_s2, kt * 48 + ntS + 16, lane);
      v16h bM = load_b_frag(Wp_s2, kt * 48 + ntS + 32, lane);
      aS0 = wmma_f16(a0, bS, aS0);
      aS1 = wmma_f16(a1, bS, aS1);
      aE0 = wmma_f16(a0, bE, aE0);
      aE1 = wmma_f16(a1, bE, aE1);
      aM0 = wmma_f16(a0, bM, aM0);
      aM1 = wmma_f16(a1, bM, aM1);
    }
    int colS = cb + n;
    float bS = b_s2[colS], bE = b_s2[256 + colS], bM = b_s2[512 + colS];
#pragma unroll
    for (int mt = 0; mt < 2; ++mt) {
#pragma unroll
      for (int r = 0; r < 8; ++r) {
        int row = mt * 16 + r + 8 * hi;
        int d   = shDst[row];
        float rlS = 0.f, rlE = 0.f, rlM = 0.f;
#pragma unroll
        for (int q = 0; q < RDIM; ++q) {
          float rb = shRbf[row][q];
          rlS += rb * W_rbf_lin[q * 768 + colS];
          rlE += rb * W_rbf_lin[q * 768 + 256 + colS];
          rlM += rb * W_rbf_lin[q * 768 + 512 + colS];
        }
        float hS = (mt ? aS1[r] : aS0[r]) + bS;
        float hE = (mt ? aE1[r] : aE0[r]) + bE;
        float hM = (mt ? aM1[r] : aM0[r]) + bM;
        float fS = hS * rlS;   // g_state
        float fE = hE * rlE;   // g_edge
        float fM = hM * rlM;   // ms
        size_t off = (size_t)d * FDIM + colS;
        atomicAdd(&sum_ms[off],   fM);
        atomicAdd(&sumsq_ms[off], fM * fM);
        unsigned u = enc_f(fM);
        atomicMin(&minU[off], u);
        atomicMax(&maxU[off], u);
        float mv0 = x[off] * fS;   // x[dst][c]*g_state, broadcast over k
#pragma unroll
        for (int k = 0; k < 3; ++k)
          atomicAdd(&out_v[(size_t)d * 3 * FDIM + k * FDIM + colS],
                    mv0 + shEv[row][k] * fE);
      }
    }
  }
}

// ---------------------------------------------------------------------------
// Fused node kernel: 16 nodes per block. Builds 16x3328 f16 A in dynamic LDS
// [x | agg(4F) | amp(4F) | att(4F)], then WMMA against packed W_post.
// ---------------------------------------------------------------------------
__global__ __launch_bounds__(256) void node_kernel(
    const float* __restrict__ x,
    const float* __restrict__ deg,
    const float* __restrict__ sum_ms,
    const float* __restrict__ sumsq_ms,
    const unsigned* __restrict__ minU,
    const unsigned* __restrict__ maxU,
    const float* __restrict__ logsum,
    const _Float16* __restrict__ Wp_post,
    const float* __restrict__ b_post,
    float* __restrict__ out_x) {
  extern __shared__ __align__(16) _Float16 shDynN[];
  _Float16(*A)[3328] = reinterpret_cast<_Float16(*)[3328]>(shDynN);

  const int tid  = threadIdx.x;
  const int lane = tid & 31;
  const int wave = tid >> 5;
  const int hi   = lane >> 4;
  const int n    = lane & 15;
  const int i0   = blockIdx.x * 16;
  const float avg_log = *logsum / (float)N_NODES;

  // -------- stage: statistics -> A row --------
  {
    int row  = tid & 15;
    int c0   = (tid >> 4) * 16;
    int node = i0 + row;
    float dg    = deg[node];
    float dc    = fmaxf(dg, 1.0f);
    float logd  = log1pf(dg);
    float s_amp = logd / avg_log;
    float s_att = avg_log / log1pf(dc);
    bool  has   = dg > 0.0f;
    size_t base = (size_t)node * FDIM;
#pragma unroll 4
    for (int c = 0; c < 16; ++c) {
      int col = c0 + c;
      float mean = sum_ms[base + col] / dc;
      float msq  = sumsq_ms[base + col] / dc;
      float sd   = sqrtf(fmaxf(msq - mean * mean, 0.0f) + 1e-5f);
      float mn   = has ? dec_f(minU[base + col]) : 0.0f;
      float mx   = has ? dec_f(maxU[base + col]) : 0.0f;
      A[row][col] = (_Float16)x[base + col];
      float ag[4] = {mean, mn, mx, sd};
#pragma unroll
      for (int a4 = 0; a4 < 4; ++a4) {
        A[row][256  + a4 * 256 + col] = (_Float16)ag[a4];
        A[row][1280 + a4 * 256 + col] = (_Float16)(ag[a4] * s_amp);
        A[row][2304 + a4 * 256 + col] = (_Float16)(ag[a4] * s_att);
      }
    }
  }
  __syncthreads();

  // -------- GEMM: delta_x = A(16x3328) @ W_post(3328x256) --------
  v8f acc0 = {}, acc1 = {};
  const _Float16* Arow = &A[n][0];
  for (int kt = 0; kt < 104; ++kt) {
    v16h a = load_a_frag(Arow + kt * 32, hi);
    acc0 = wmma_f16(a, load_b_frag(Wp_post, kt * 16 + 2 * wave,     lane), acc0);
    acc1 = wmma_f16(a, load_b_frag(Wp_post, kt * 16 + 2 * wave + 1, lane), acc1);
  }
#pragma unroll
  for (int t = 0; t < 2; ++t) {
    v8f acc = t ? acc1 : acc0;
    int col = wave * 32 + t * 16 + n;
    float bp = b_post[col];
#pragma unroll
    for (int r = 0; r < 8; ++r) {
      size_t off = (size_t)(i0 + r + 8 * hi) * FDIM + col;
      out_x[off] = x[off] + acc[r] + bp;
    }
  }
}

// ---------------------------------------------------------------------------
// Host side
// ---------------------------------------------------------------------------
extern "C" void kernel_launch(void* const* d_in, const int* in_sizes, int n_in,
                              void* d_out, int out_size, void* d_ws, size_t ws_size,
                              hipStream_t stream) {
  (void)in_sizes; (void)n_in; (void)out_size; (void)ws_size;
  const float* x         = (const float*)d_in[0];
  const float* v         = (const float*)d_in[1];
  const float* edge_rbf  = (const float*)d_in[2];
  const float* edge_vec  = (const float*)d_in[3];
  const float* W_rbf_emb = (const float*)d_in[4];
  const float* b_rbf_emb = (const float*)d_in[5];
  const float* W_pre     = (const float*)d_in[6];
  const float* b_pre     = (const float*)d_in[7];
  const float* W_s1      = (const float*)d_in[8];
  const float* b_s1      = (const float*)d_in[9];
  const float* W_s2      = (const float*)d_in[10];
  const float* b_s2      = (const float*)d_in[11];
  const float* W_rbf_lin = (const float*)d_in[12];
  const float* W_post    = (const float*)d_in[13];
  const float* b_post    = (const float*)d_in[14];
  const int*   eidx      = (const int*)d_in[15];
  const int* src = eidx;
  const int* dst = eidx + N_EDGES;

  float* out_x = (float*)d_out;
  float* out_v = out_x + (size_t)N_NODES * FDIM;

  // ---- workspace layout (256B aligned) ----
  char* ws = (char*)d_ws;
  size_t off = 0;
  auto alloc = [&](size_t bytes) {
    size_t p = off;
    off += (bytes + 255) & ~(size_t)255;
    return p;
  };
  _Float16* wp_pre  = (_Float16*)(ws + alloc((size_t)768  * 256 * 2));
  _Float16* wp_s1   = (_Float16*)(ws + alloc((size_t)256  * 256 * 2));
  _Float16* wp_s2   = (_Float16*)(ws + alloc((size_t)256  * 768 * 2));
  _Float16* wp_post = (_Float16*)(ws + alloc((size_t)3328 * 256 * 2));
  float*    deg     = (float*)   (ws + alloc((size_t)N_NODES * 4));
  float*    sum_ms  = (float*)   (ws + alloc((size_t)N_NODES * FDIM * 4));
  float*    sumsq   = (float*)   (ws + alloc((size_t)N_NODES * FDIM * 4));
  unsigned* minU    = (unsigned*)(ws + alloc((size_t)N_NODES * FDIM * 4));
  unsigned* maxU    = (unsigned*)(ws + alloc((size_t)N_NODES * FDIM * 4));
  float*    logsum  = (float*)   (ws + alloc(256));

  // ---- init accumulators (graph-capture-safe async ops only) ----
  hipMemsetAsync(deg,    0,    (size_t)N_NODES * 4,        stream);
  hipMemsetAsync(sum_ms, 0,    (size_t)N_NODES * FDIM * 4, stream);
  hipMemsetAsync(sumsq,  0,    (size_t)N_NODES * FDIM * 4, stream);
  hipMemsetAsync(minU,   0xFF, (size_t)N_NODES * FDIM * 4, stream); // encoded +max
  hipMemsetAsync(maxU,   0x00, (size_t)N_NODES * FDIM * 4, stream); // encoded -max
  hipMemsetAsync(logsum, 0,    4,                          stream);
  hipMemcpyAsync(out_v, v, (size_t)N_NODES * 3 * FDIM * 4,
                 hipMemcpyDeviceToDevice, stream);

  // ---- pack weights into WMMA fragment order ----
  auto packLaunch = [&](const float* W, _Float16* O, int K, int N) {
    int total = K * N;
    pack_weight_f16<<<(total + 255) / 256, 256, 0, stream>>>(W, O, K, N);
  };
  packLaunch(W_pre,  wp_pre,  768,  256);
  packLaunch(W_s1,   wp_s1,   256,  256);
  packLaunch(W_s2,   wp_s2,   256,  768);
  packLaunch(W_post, wp_post, 3328, 256);

  // ---- degree ----
  deg_kernel<<<(N_EDGES + 255) / 256, 256, 0, stream>>>(dst, deg);

  // ---- fused edge pipeline (32 edges / block, 64 KB dynamic LDS) ----
  size_t edge_lds = (size_t)(EPB * 768 + EPB * 256) * sizeof(_Float16);
  edge_kernel<<<N_EDGES / EPB, 256, edge_lds, stream>>>(
      x, edge_rbf, edge_vec, W_rbf_emb, b_rbf_emb,
      wp_pre, b_pre, wp_s1, b_s1, wp_s2, b_s2, W_rbf_lin,
      src, dst, sum_ms, sumsq, minU, maxU, out_v);

  // ---- mean log1p(deg) ----
  avg_log_kernel<<<(N_NODES + 255) / 256, 256, 0, stream>>>(deg, logsum);

  // ---- fused node pipeline (16 nodes / block, 104 KB dynamic LDS) ----
  size_t node_lds = (size_t)16 * 3328 * sizeof(_Float16);
  node_kernel<<<N_NODES / 16, 256, node_lds, stream>>>(
      x, deg, sum_ms, sumsq, minU, maxU, logsum, wp_post, b_post, out_x);
}
